// Iterative_13460427506006
// MI455X (gfx1250) — compile-verified, matrix operand order checked by reading, not run on hardware
//
#include <hip/hip_runtime.h>
#include <hip/hip_bf16.h>

typedef __attribute__((ext_vector_type(16))) _Float16 v16h;
typedef __attribute__((ext_vector_type(8)))  float    v8f;
typedef __attribute__((ext_vector_type(4)))  unsigned int u32x4;
typedef __attribute__((ext_vector_type(4)))  float        f32x4;
typedef __attribute__((ext_vector_type(8)))  int          i32x8;
typedef __attribute__((ext_vector_type(4)))  int          i32x4;

#define LDP 40  // LDS row stride in halves (32 + 8 pad; 80B keeps 16B-aligned rows)

#if __has_builtin(__builtin_amdgcn_tensor_load_to_lds)
#define HAVE_TDM 1
#else
#define HAVE_TDM 0
#endif

// Raw 32-bit LDS byte offset of a __shared__ pointer (generic -> AS(3) cast).
__device__ __forceinline__ unsigned lds_off(const void* p) {
    return (unsigned)(unsigned long long)(__attribute__((address_space(3))) const void*)p;
}

#if HAVE_TDM
// Issue a 2D TDM tile load: `rows` rows of `cols` f16 from row-major global
// (row stride `gstride` elements) into LDS at `ldsByte`.  If `padded`, the TDM
// inserts 4 dwords of LDS padding after every 16 dwords (one 32-half row),
// producing exactly the LDP=40 layout.  EXEC-independent; one issue per wave.
__device__ __forceinline__ void tdm_load_2d(const _Float16* gsrc, unsigned ldsByte,
                                            unsigned cols, unsigned rows,
                                            unsigned gstride, bool padded) {
    unsigned long long ga = (unsigned long long)gsrc;
    u32x4 g0;
    g0[0] = 1u;                                   // count=1, user mode
    g0[1] = ldsByte;                              // lds_addr (bytes)
    g0[2] = (unsigned)ga;                         // global_addr[31:0]
    g0[3] = (unsigned)(ga >> 32) | (2u << 30);    // global_addr[56:32] | type=2
    i32x8 g1;
    unsigned d0 = (1u << 16);                     // data_size = 2B
    if (padded) d0 |= (1u << 20) | (3u << 22) | (3u << 25);  // pad: every 16dw, +4dw
    g1[0] = (int)d0;
    g1[1] = (int)(cols << 16);    // tensor_dim0[15:0]
    g1[2] = (int)(rows << 16);    // tensor_dim0[31:16]=0 | tensor_dim1[15:0]
    g1[3] = (int)(cols << 16);    // tensor_dim1[31:16]=0 | tile_dim0
    g1[4] = (int)rows;            // tile_dim1 | tile_dim2=0
    g1[5] = (int)gstride;         // tensor_dim0_stride[31:0]
    g1[6] = 0;
    g1[7] = 0;
    i32x4 z4 = {0, 0, 0, 0};
#if __clang_major__ >= 23
    i32x8 z8 = {0, 0, 0, 0, 0, 0, 0, 0};
    __builtin_amdgcn_tensor_load_to_lds(g0, g1, z4, z4, z8, 0);
#else
    __builtin_amdgcn_tensor_load_to_lds(g0, g1, z4, z4, 0);
#endif
}
#endif

// Assemble a 16-bit WMMA A/B fragment from an LDS tile stored [row][k] with
// row stride `lda` halves. Lane L owns row (rowBase + L%16); per ISA 7.12.2,
// VGPR v holds K pair at k = (v/4)*16 + (L/16)*8 + (v%4)*2.
__device__ __forceinline__ v16h frag_ld(const _Float16* tile, int lda, int rowBase, int lane) {
    const _Float16* row = tile + (size_t)(rowBase + (lane & 15)) * lda;
    const int half = lane >> 4;
    v16h f;
#pragma unroll
    for (int v = 0; v < 8; ++v) {
        int k = ((v >> 2) << 4) + (half << 3) + ((v & 3) << 1);
        f[2 * v]     = row[k];
        f[2 * v + 1] = row[k + 1];
    }
    return f;
}

#define WMMA_F16(A, B, C) \
    __builtin_amdgcn_wmma_f32_16x16x32_f16(false, (A), false, (B), (short)0, (C), false, false)

__global__ void zero_f32(float* p, int n) {
    int i = blockIdx.x * blockDim.x + threadIdx.x;
    if (i < n) p[i] = 0.f;
}

__global__ void cvt_reduce_w(const float* w, _Float16* w16, int n) {
    int i = blockIdx.x * blockDim.x + threadIdx.x;
    if (i < n) w16[i] = (_Float16)w[i];
}

// conv3_w [O=512][C=512][3][3] -> wk16 [kidx=kh*3+kw][O][C] f16
__global__ void cvt_conv_w(const float* w, _Float16* wk) {
    int i = blockIdx.x * blockDim.x + threadIdx.x;
    if (i >= 9 * 512 * 512) return;
    int kidx = i / (512 * 512);
    int rem  = i - kidx * 512 * 512;
    int o = rem >> 9, c = rem & 511;
    int kh = kidx / 3, kw = kidx % 3;
    wk[i] = (_Float16)w[(((size_t)o * 512 + c) * 3 + kh) * 3 + kw];
}

// ---------------------------------------------------------------------------
// Kernel 1: reduce 1x1 conv.  out[o,p] = sum_c W[o,c] * feat[b,c,p]
//   128 thr (4 waves); tile 128x32; double-buffered LDS, 1 barrier / K-step
//   fused +bias, BN, ReLU -> xf16 [b][512][224] (cols 196..223 zeroed)
// ---------------------------------------------------------------------------
__global__ __launch_bounds__(128) void reduce_gemm(
    const float* __restrict__ feat, const _Float16* __restrict__ w16,
    const float* __restrict__ rb, const float* __restrict__ g,
    const float* __restrict__ be, const float* __restrict__ mn,
    const float* __restrict__ vr, _Float16* __restrict__ xf16) {
    const int b = blockIdx.z, m0 = blockIdx.y * 128, n0 = blockIdx.x * 32;
    const int tid = threadIdx.x, lane = tid & 31, wv = tid >> 5;
    __shared__ _Float16 As[2][128 * LDP];
    __shared__ _Float16 Bs[2][32 * LDP];
    v8f c00 = {}, c01 = {}, c10 = {}, c11 = {};
    const float* B = feat + (size_t)b * 2048 * 196;
    const bool full = (n0 + 32 <= 196);   // block-uniform

    auto stageA = [&](int buf, int k0) {
        const u32x4* src = (const u32x4*)(w16 + (size_t)(m0 + tid) * 2048 + k0);
        u32x4* dst = (u32x4*)(&As[buf][tid * LDP]);
#pragma unroll
        for (int i = 0; i < 4; ++i) dst[i] = src[i];
    };
    auto stageB = [&](int buf, int k0) {
        int kk = tid >> 2, n8 = (tid & 3) * 8;
        const float* src = B + (size_t)(k0 + kk) * 196 + n0 + n8;
        _Float16* dst = &Bs[buf][0];
        if (full) {
            f32x4 v0 = ((const f32x4*)src)[0];
            f32x4 v1 = ((const f32x4*)src)[1];
#pragma unroll
            for (int i = 0; i < 4; ++i) dst[(n8 + i) * LDP + kk]     = (_Float16)v0[i];
#pragma unroll
            for (int i = 0; i < 4; ++i) dst[(n8 + 4 + i) * LDP + kk] = (_Float16)v1[i];
        } else {
#pragma unroll
            for (int i = 0; i < 8; ++i) {
                float v = 0.f;
                if (n0 + n8 + i < 196) v = src[i];
                dst[(n8 + i) * LDP + kk] = (_Float16)v;
            }
        }
    };

    stageA(0, 0);
    stageB(0, 0);
    __syncthreads();
    for (int k0 = 0; k0 < 2048; k0 += 32) {
        const int cur = (k0 >> 5) & 1, nxt = cur ^ 1;
        if (k0 + 32 < 2048) {
            stageA(nxt, k0 + 32);
            stageB(nxt, k0 + 32);
            __builtin_prefetch(B + (size_t)(k0 + 64) * 196 + n0, 0, 1);
        }
        v16h a0 = frag_ld(&As[cur][0], LDP, wv * 32,      lane);
        v16h a1 = frag_ld(&As[cur][0], LDP, wv * 32 + 16, lane);
        v16h b0 = frag_ld(&Bs[cur][0], LDP, 0,  lane);
        v16h b1 = frag_ld(&Bs[cur][0], LDP, 16, lane);
        c00 = WMMA_F16(a0, b0, c00);
        c01 = WMMA_F16(a0, b1, c01);
        c10 = WMMA_F16(a1, b0, c10);
        c11 = WMMA_F16(a1, b1, c11);
        __syncthreads();
    }
    const int half = lane >> 4, nl = lane & 15;
    const int p0 = n0 + nl, p1 = n0 + 16 + nl;
#pragma unroll
    for (int r = 0; r < 8; ++r) {
#pragma unroll
        for (int sub = 0; sub < 2; ++sub) {
            int o = m0 + wv * 32 + sub * 16 + r + 8 * half;
            float inv = g[o] * rsqrtf(vr[o] + 1e-5f);
            float sh  = be[o] - mn[o] * inv;
            float x0 = sub ? c10[r] : c00[r];
            float x1 = sub ? c11[r] : c01[r];
            float v0 = fmaxf((x0 + rb[o]) * inv + sh, 0.f);
            float v1 = fmaxf((x1 + rb[o]) * inv + sh, 0.f);
            size_t base = ((size_t)b * 512 + o) * 224;
            xf16[base + p0] = (_Float16)((p0 < 196) ? v0 : 0.f);
            xf16[base + p1] = (_Float16)((p1 < 196) ? v1 : 0.f);
        }
    }
}

// ---------------------------------------------------------------------------
// Kernel 2: xxt[i,j] = sum_p xf[i,p] xf[j,p]   (K = 224 padded, pads are 0)
//   A block [128 x 224] resident in LDS via one TDM load; Bs double-buffered.
// ---------------------------------------------------------------------------
__global__ __launch_bounds__(128) void xxt_gemm(
    const _Float16* __restrict__ xf16, float* __restrict__ xxt) {
    const int b = blockIdx.z, m0 = blockIdx.y * 128, n0 = blockIdx.x * 32;
    const int tid = threadIdx.x, lane = tid & 31, wv = tid >> 5;
    __shared__ _Float16 At[128 * 224];
    __shared__ _Float16 Bs[2][32 * LDP];
    v8f c00 = {}, c01 = {}, c10 = {}, c11 = {};
    const _Float16* X = xf16 + (size_t)b * 512 * 224;

    auto stageB = [&](int buf, int k0) {
        int n = tid >> 2, k8 = (tid & 3) * 8;
        const u32x4* src = (const u32x4*)(X + (size_t)(n0 + n) * 224 + k0 + k8);
        *(u32x4*)(&Bs[buf][n * LDP + k8]) = *src;
    };

#if HAVE_TDM
    if (wv == 0) {
        tdm_load_2d(X + (size_t)m0 * 224, lds_off(At), 224u, 128u, 224u, false);
    }
#else
    {
        const u32x4* s = (const u32x4*)(X + (size_t)(m0 + tid) * 224);
        u32x4* d = (u32x4*)(At + (size_t)tid * 224);
#pragma unroll
        for (int i = 0; i < 28; ++i) d[i] = s[i];
    }
#endif
    stageB(0, 0);
#if HAVE_TDM
    if (wv == 0) __builtin_amdgcn_s_wait_tensorcnt((short)0);
#endif
    __syncthreads();

    for (int k0 = 0; k0 < 224; k0 += 32) {
        const int cur = (k0 >> 5) & 1, nxt = cur ^ 1;
        if (k0 + 32 < 224) stageB(nxt, k0 + 32);
        v16h a0 = frag_ld(At + k0, 224, wv * 32,      lane);
        v16h a1 = frag_ld(At + k0, 224, wv * 32 + 16, lane);
        v16h b0 = frag_ld(&Bs[cur][0], LDP, 0,  lane);
        v16h b1 = frag_ld(&Bs[cur][0], LDP, 16, lane);
        c00 = WMMA_F16(a0, b0, c00);
        c01 = WMMA_F16(a0, b1, c01);
        c10 = WMMA_F16(a1, b0, c10);
        c11 = WMMA_F16(a1, b1, c11);
        __syncthreads();
    }
    const int half = lane >> 4, nl = lane & 15;
#pragma unroll
    for (int r = 0; r < 8; ++r) {
#pragma unroll
        for (int sub = 0; sub < 2; ++sub) {
            int i = m0 + wv * 32 + sub * 16 + r + 8 * half;
            size_t base = ((size_t)b * 512 + i) * 512;
            xxt[base + n0 + nl]      = sub ? c10[r] : c00[r];
            xxt[base + n0 + 16 + nl] = sub ? c11[r] : c01[r];
        }
    }
}

// ---------------------------------------------------------------------------
// Kernel 3: att = softmax(-xxt) along rows of 512, stored f16
// ---------------------------------------------------------------------------
__global__ __launch_bounds__(256) void softmax_neg(
    const float* __restrict__ xxt, _Float16* __restrict__ att) {
    const size_t row = blockIdx.x;
    const float* x = xxt + row * 512;
    _Float16* a = att + row * 512;
    __shared__ float sm[256];
    int t = threadIdx.x;
    float v0 = -x[t], v1 = -x[t + 256];
    sm[t] = fmaxf(v0, v1);
    __syncthreads();
    for (int s = 128; s > 0; s >>= 1) { if (t < s) sm[t] = fmaxf(sm[t], sm[t + s]); __syncthreads(); }
    float mx = sm[0];
    __syncthreads();
    float e0 = __expf(v0 - mx), e1 = __expf(v1 - mx);
    sm[t] = e0 + e1;
    __syncthreads();
    for (int s = 128; s > 0; s >>= 1) { if (t < s) sm[t] += sm[t + s]; __syncthreads(); }
    float inv = 1.f / sm[0];
    a[t]       = (_Float16)(e0 * inv);
    a[t + 256] = (_Float16)(e1 * inv);
}

// ---------------------------------------------------------------------------
// Kernel 4: y[c,p] = sum_j att[c,j] xf[j,p]   (K = 512) -> y16 [b][512][224]
//   double-buffered LDS, one barrier per K-step
// ---------------------------------------------------------------------------
__global__ __launch_bounds__(128) void y_gemm(
    const _Float16* __restrict__ att16, const _Float16* __restrict__ xf16,
    _Float16* __restrict__ y16) {
    const int b = blockIdx.z, m0 = blockIdx.y * 128, n0 = blockIdx.x * 32;
    const int tid = threadIdx.x, lane = tid & 31, wv = tid >> 5;
    __shared__ _Float16 As[2][128 * LDP];
    __shared__ _Float16 Bs[2][32 * LDP];
    v8f c00 = {}, c01 = {}, c10 = {}, c11 = {};
    const _Float16* A = att16 + (size_t)b * 512 * 512;
    const _Float16* X = xf16  + (size_t)b * 512 * 224;

    auto stageA = [&](int buf, int k0) {
        const u32x4* src = (const u32x4*)(A + (size_t)(m0 + tid) * 512 + k0);
        u32x4* dst = (u32x4*)(&As[buf][tid * LDP]);
#pragma unroll
        for (int i = 0; i < 4; ++i) dst[i] = src[i];
    };
    auto stageB = [&](int buf, int k0) {
        int kk = tid >> 2, n8 = (tid & 3) * 8;
        const _Float16* src = X + (size_t)(k0 + kk) * 224 + n0 + n8;
        u32x4 raw = *(const u32x4*)src;
        const _Float16* h = (const _Float16*)&raw;
#pragma unroll
        for (int i = 0; i < 8; ++i) Bs[buf][(n8 + i) * LDP + kk] = h[i];
    };

    stageA(0, 0);
    stageB(0, 0);
    __syncthreads();
    for (int k0 = 0; k0 < 512; k0 += 32) {
        const int cur = (k0 >> 5) & 1, nxt = cur ^ 1;
        if (k0 + 32 < 512) {
            stageA(nxt, k0 + 32);
            stageB(nxt, k0 + 32);
            __builtin_prefetch(X + (size_t)(k0 + 64) * 224 + n0, 0, 1);
        }
        v16h a0 = frag_ld(&As[cur][0], LDP, wv * 32,      lane);
        v16h a1 = frag_ld(&As[cur][0], LDP, wv * 32 + 16, lane);
        v16h b0 = frag_ld(&Bs[cur][0], LDP, 0,  lane);
        v16h b1 = frag_ld(&Bs[cur][0], LDP, 16, lane);
        c00 = WMMA_F16(a0, b0, c00);
        c01 = WMMA_F16(a0, b1, c01);
        c10 = WMMA_F16(a1, b0, c10);
        c11 = WMMA_F16(a1, b1, c11);
        __syncthreads();
    }
    const int half = lane >> 4, nl = lane & 15;
    const int p0 = n0 + nl, p1 = n0 + 16 + nl;
#pragma unroll
    for (int r = 0; r < 8; ++r) {
#pragma unroll
        for (int sub = 0; sub < 2; ++sub) {
            int c = m0 + wv * 32 + sub * 16 + r + 8 * half;
            float x0 = sub ? c10[r] : c00[r];
            float x1 = sub ? c11[r] : c01[r];
            size_t base = ((size_t)b * 512 + c) * 224;
            y16[base + p0] = (_Float16)((p0 < 196) ? x0 : 0.f);
            y16[base + p1] = (_Float16)((p1 < 196) ? x1 : 0.f);
        }
    }
}

// ---------------------------------------------------------------------------
// Kernel 5: 3x3 conv as 9-phase implicit GEMM (K = 9*512).
//   A-tiles DMA'd by TDM into double-buffered LDS (TENSORcnt pipelined, TDM
//   pad reproduces the LDP=40 layout); fused bias + BN + ReLU + avg-pool.
// ---------------------------------------------------------------------------
__global__ __launch_bounds__(128) void conv_gemm(
    const _Float16* __restrict__ y16, const _Float16* __restrict__ wk16,
    const float* __restrict__ cb, const float* __restrict__ g,
    const float* __restrict__ be, const float* __restrict__ mn,
    const float* __restrict__ vr, float* __restrict__ z) {
    const int b = blockIdx.z, m0 = blockIdx.y * 128, n0 = blockIdx.x * 32;
    const int tid = threadIdx.x, lane = tid & 31, wv = tid >> 5;
    __shared__ _Float16 As[2][128 * LDP];
    __shared__ _Float16 Bs[2][32 * LDP];
    __shared__ float red[128 * 33];
    v8f c00 = {}, c01 = {}, c10 = {}, c11 = {};
    const _Float16* Y = y16 + (size_t)b * 512 * 224;

    auto issueA = [&](int buf, int kk) {
        const int kidx = kk >> 4, cbase = (kk & 15) * 32;
        const _Float16* src = wk16 + ((size_t)kidx * 512 + m0) * 512 + cbase;
#if HAVE_TDM
        if (wv == 0) tdm_load_2d(src, lds_off(&As[buf][0]), 32u, 128u, 512u, true);
#else
        const u32x4* s = (const u32x4*)(src + (size_t)tid * 512);
        u32x4* d = (u32x4*)(&As[buf][tid * LDP]);
#pragma unroll
        for (int i = 0; i < 4; ++i) d[i] = s[i];
#endif
    };
    auto stageB = [&](int buf, int kk) {
        const int kidx = kk >> 4, cbase = (kk & 15) * 32;
        const int dh = kidx / 3 - 1, dw = kidx % 3 - 1;
        int kx = tid >> 2, n8 = (tid & 3) * 8;
        const _Float16* yrow = Y + (size_t)(cbase + kx) * 224;
#pragma unroll
        for (int i = 0; i < 8; ++i) {
            int p = n0 + n8 + i;
            _Float16 v = (_Float16)0.f;
            if (p < 196) {
                int h = p / 14 + dh, w = p % 14 + dw;
                if ((unsigned)h < 14u && (unsigned)w < 14u) v = yrow[h * 14 + w];
            }
            Bs[buf][(n8 + i) * LDP + kx] = v;
        }
    };

    issueA(0, 0);
    stageB(0, 0);
#if HAVE_TDM
    if (wv == 0) __builtin_amdgcn_s_wait_tensorcnt((short)0);
#endif
    __syncthreads();

    for (int kk = 0; kk < 144; ++kk) {
        const int cur = kk & 1, nxt = cur ^ 1;
        if (kk + 1 < 144) {
            issueA(nxt, kk + 1);   // TDM runs ahead under TENSORcnt
            stageB(nxt, kk + 1);
        }
        v16h a0 = frag_ld(&As[cur][0], LDP, wv * 32,      lane);
        v16h a1 = frag_ld(&As[cur][0], LDP, wv * 32 + 16, lane);
        v16h b0 = frag_ld(&Bs[cur][0], LDP, 0,  lane);
        v16h b1 = frag_ld(&Bs[cur][0], LDP, 16, lane);
        c00 = WMMA_F16(a0, b0, c00);
        c01 = WMMA_F16(a0, b1, c01);
        c10 = WMMA_F16(a1, b0, c10);
        c11 = WMMA_F16(a1, b1, c11);
#if HAVE_TDM
        if (wv == 0) __builtin_amdgcn_s_wait_tensorcnt((short)0);
#endif
        __syncthreads();
    }
    const int half = lane >> 4, nl = lane & 15;
#pragma unroll
    for (int r = 0; r < 8; ++r) {
#pragma unroll
        for (int sub = 0; sub < 2; ++sub) {
            int mr = wv * 32 + sub * 16 + r + 8 * half;
            int o  = m0 + mr;
            float inv = g[o] * rsqrtf(vr[o] + 1e-5f);
            float sh  = be[o] - mn[o] * inv;
            float x0 = sub ? c10[r] : c00[r];
            float x1 = sub ? c11[r] : c01[r];
            float v0 = fmaxf((x0 + cb[o]) * inv + sh, 0.f);
            float v1 = fmaxf((x1 + cb[o]) * inv + sh, 0.f);
            red[mr * 33 + nl]      = (n0 + nl      < 196) ? v0 : 0.f;
            red[mr * 33 + 16 + nl] = (n0 + 16 + nl < 196) ? v1 : 0.f;
        }
    }
    __syncthreads();
    {
        float s = 0.f;
#pragma unroll
        for (int j = 0; j < 32; ++j) s += red[tid * 33 + j];
        atomicAdd(z + (size_t)b * 512 + m0 + tid, s * (1.f / 196.f));
    }
}

// ---------------------------------------------------------------------------
// Kernel 6: FC stack, accumulates branch result into d_out
// ---------------------------------------------------------------------------
__global__ __launch_bounds__(256) void fc_accum(
    const float* __restrict__ z, const float* __restrict__ fc1w,
    const float* __restrict__ fc1b, const float* __restrict__ fc2w,
    const float* __restrict__ fc2b, float* __restrict__ out) {
    const int b = blockIdx.x, t = threadIdx.x;
    __shared__ float zs[512];
    __shared__ float hs[200];
    zs[t]       = z[(size_t)b * 512 + t];
    zs[t + 256] = z[(size_t)b * 512 + t + 256];
    __syncthreads();
    if (t < 200) {
        float acc = fc1b[t];
        for (int k = 0; k < 512; ++k) acc += zs[k] * fc1w[(size_t)t * 512 + k];
        hs[t] = fmaxf(acc, 0.f);
    }
    __syncthreads();
    if (t < 200) {
        float acc = fc2b[t];
        for (int j = 0; j < 200; ++j) acc += hs[j] * fc2w[(size_t)t * 200 + j];
        out[(size_t)b * 200 + t] += acc;
    }
}

// ---------------------------------------------------------------------------
extern "C" void kernel_launch(void* const* d_in, const int* in_sizes, int n_in,
                              void* d_out, int out_size, void* d_ws, size_t ws_size,
                              hipStream_t stream) {
    const float* feature[2] = { (const float*)d_in[0], (const float*)d_in[1] };
    const float* reduce_w   = (const float*)d_in[2];
    const float* reduce_b   = (const float*)d_in[3];
    const float* reduce_g   = (const float*)d_in[4];
    const float* reduce_be  = (const float*)d_in[5];
    const float* reduce_mn  = (const float*)d_in[6];
    const float* reduce_vr  = (const float*)d_in[7];
    const float* conv3_w    = (const float*)d_in[8];
    const float* conv3_b    = (const float*)d_in[9];
    const float* conv3_g    = (const float*)d_in[10];
    const float* conv3_be   = (const float*)d_in[11];
    const float* conv3_mn   = (const float*)d_in[12];
    const float* conv3_vr   = (const float*)d_in[13];
    const float* fc1w = (const float*)d_in[14];
    const float* fc1b = (const float*)d_in[15];
    const float* fc2w = (const float*)d_in[16];
    const float* fc2b = (const float*)d_in[17];

    char* ws = (char*)d_ws;
    const size_t OFF_W16  = 0;                                       // 512*2048 f16
    const size_t OFF_WK16 = OFF_W16  + (size_t)512 * 2048 * 2;       // 9*512*512 f16
    const size_t OFF_XF   = OFF_WK16 + (size_t)9 * 512 * 512 * 2;    // 64*512*224 f16
    const size_t OFF_XXT  = OFF_XF   + (size_t)64 * 512 * 224 * 2;   // 64*512*512 f32
    const size_t OFF_ATT  = OFF_XXT  + (size_t)64 * 512 * 512 * 4;   // 64*512*512 f16
    const size_t OFF_Y    = OFF_ATT  + (size_t)64 * 512 * 512 * 2;   // 64*512*224 f16
    const size_t OFF_Z    = OFF_Y    + (size_t)64 * 512 * 224 * 2;   // 64*512 f32

    _Float16* w16  = (_Float16*)(ws + OFF_W16);
    _Float16* wk16 = (_Float16*)(ws + OFF_WK16);
    _Float16* xf16 = (_Float16*)(ws + OFF_XF);
    float*    xxt  = (float*)   (ws + OFF_XXT);
    _Float16* att  = (_Float16*)(ws + OFF_ATT);
    _Float16* y16  = (_Float16*)(ws + OFF_Y);
    float*    zbuf = (float*)   (ws + OFF_Z);
    float*    out  = (float*)d_out;

    zero_f32<<<(64 * 200 + 255) / 256, 256, 0, stream>>>(out, 64 * 200);
    cvt_reduce_w<<<(512 * 2048 + 255) / 256, 256, 0, stream>>>(reduce_w, w16, 512 * 2048);
    cvt_conv_w<<<(9 * 512 * 512 + 255) / 256, 256, 0, stream>>>(conv3_w, wk16);

    for (int br = 0; br < 2; ++br) {
        zero_f32<<<(64 * 512 + 255) / 256, 256, 0, stream>>>(zbuf, 64 * 512);
        reduce_gemm<<<dim3(7, 4, 64), 128, 0, stream>>>(
            feature[br], w16, reduce_b, reduce_g, reduce_be, reduce_mn, reduce_vr, xf16);
        xxt_gemm<<<dim3(16, 4, 64), 128, 0, stream>>>(xf16, xxt);
        softmax_neg<<<64 * 512, 256, 0, stream>>>(xxt, att);
        y_gemm<<<dim3(7, 4, 64), 128, 0, stream>>>(att, xf16, y16);
        conv_gemm<<<dim3(7, 4, 64), 128, 0, stream>>>(
            y16, wk16, conv3_b, conv3_g, conv3_be, conv3_mn, conv3_vr, zbuf);
        fc_accum<<<64, 256, 0, stream>>>(zbuf, fc1w, fc1b, fc2w, fc2b, out);
    }
    (void)in_sizes; (void)n_in; (void)out_size; (void)ws_size;
}
